// FastAttention_83416854823227
// MI455X (gfx1250) — compile-verified, matrix operand order checked by reading, not run on hardware
//
#include <hip/hip_runtime.h>
#include <hip/hip_bf16.h>

#define H_ 16
#define N_ 4096
#define D_ 64
#define M_ 256
#define EPS_ 1e-4f
#define DN_ 0.3535533905932738f   /* 64^-0.25 */
#define RATIO_ 0.0625f            /* 256^-0.5 */

typedef __bf16 bf16;
typedef __attribute__((ext_vector_type(16))) __bf16 v16bf;
typedef __attribute__((ext_vector_type(8)))  float  v8f;

// ---------------------------------------------------------------- utilities
__device__ inline void atomicMaxFloat(float* addr, float val) {
    if (!(val == val)) return;
    if (val >= 0.0f) atomicMax((int*)addr, __float_as_int(val));
    else             atomicMin((unsigned int*)addr, (unsigned int)__float_as_int(val));
}

// Load one 16x32 bf16 A/B fragment row chunk for this lane.
// p points at row_base + kc (row is M/N index == lane&15 handled by caller).
// Lane<16 takes K elements {0..7, 16..23}; lane>=16 takes {8..15, 24..31}.
__device__ inline v16bf load_row_frag(const bf16* p, int lane) {
    union { v16bf v; uint4 u[2]; } f;
    const int off = (lane & 16) ? 8 : 0;
    f.u[0] = *(const uint4*)(p + off);
    f.u[1] = *(const uint4*)(p + 16 + off);
    return f.v;
}

// ---------------------------------------------------------------- init
__global__ void init_kernel(float* ctx, float* ksum, float* gmax) {
    size_t i = (size_t)blockIdx.x * 256 + threadIdx.x;
    if (i < (size_t)H_ * M_ * D_) ctx[i] = 0.0f;
    if (i < (size_t)H_ * M_)      ksum[i] = 0.0f;
    if (i == 0)                   *gmax = -__builtin_inff();
}

// ------------------------------------------------- pass 1: global max of k-dash
__global__ void kdash_max_kernel(const float* __restrict__ k,
                                 const float* __restrict__ proj,
                                 float* gmax) {
    __shared__ float rows[64][64];
    __shared__ float pr[64][65];
    __shared__ float red[256];
    const int h = blockIdx.y;
    const int r0 = blockIdx.x * 64;
    const int t = threadIdx.x;

    for (int i = 0; i < 16; ++i) {
        int idx = t + i * 256; int rr = idx >> 6, d = idx & 63;
        rows[rr][d] = DN_ * k[((size_t)h * N_ + r0 + rr) * D_ + d];
    }
    float lmax = -__builtin_inff();
    const int ml = t & 63, rg = t >> 6;
    for (int c = 0; c < 4; ++c) {
        __syncthreads();
        for (int i = 0; i < 16; ++i) {
            int idx = t + i * 256; int mm = idx >> 6, d = idx & 63;
            pr[mm][d] = proj[(size_t)(c * 64 + mm) * D_ + d];
        }
        __syncthreads();
        for (int rr = rg * 16; rr < rg * 16 + 16; ++rr) {
            float s = 0.0f;
#pragma unroll
            for (int d = 0; d < 64; ++d) s += rows[rr][d] * pr[ml][d];
            lmax = fmaxf(lmax, s);
        }
    }
    red[t] = lmax; __syncthreads();
    for (int off = 128; off > 0; off >>= 1) {
        if (t < off) red[t] = fmaxf(red[t], red[t + off]);
        __syncthreads();
    }
    if (t == 0) atomicMaxFloat(gmax, red[0]);
}

// ------------------------------------------------- pass 2: kp features + ksum
__global__ void kp_kernel(const float* __restrict__ k,
                          const float* __restrict__ proj,
                          const float* __restrict__ gmax,
                          bf16* __restrict__ kp, float* ksum) {
    __shared__ float rows[64][64];
    __shared__ float pr[64][65];
    __shared__ float diag[64];
    __shared__ float msum[64];
    const int h = blockIdx.y;
    const int r0 = blockIdx.x * 64;
    const int t = threadIdx.x;

    for (int i = 0; i < 16; ++i) {
        int idx = t + i * 256; int rr = idx >> 6, d = idx & 63;
        rows[rr][d] = DN_ * k[((size_t)h * N_ + r0 + rr) * D_ + d];
    }
    __syncthreads();
    if (t < 64) {
        float s = 0.0f;
#pragma unroll
        for (int d = 0; d < 64; ++d) s += rows[t][d] * rows[t][d];
        diag[t] = 0.5f * s;   // == 0.5*dn^2*sum(raw^2) since rows pre-scaled by dn
    }
    const float gm = *gmax;
    const int ml = t & 63, rg = t >> 6;
    for (int c = 0; c < 4; ++c) {
        __syncthreads();
        for (int i = 0; i < 16; ++i) {
            int idx = t + i * 256; int mm = idx >> 6, d = idx & 63;
            pr[mm][d] = proj[(size_t)(c * 64 + mm) * D_ + d];
        }
        if (t < 64) msum[t] = 0.0f;
        __syncthreads();
        float part = 0.0f;
        for (int rr = rg * 16; rr < rg * 16 + 16; ++rr) {
            float s = 0.0f;
#pragma unroll
            for (int d = 0; d < 64; ++d) s += rows[rr][d] * pr[ml][d];
            float val = RATIO_ * (__expf(s - diag[rr] - gm) + EPS_);
            kp[((size_t)h * N_ + r0 + rr) * M_ + c * 64 + ml] = (bf16)val;
            part += val;
        }
        atomicAdd(&msum[ml], part);
        __syncthreads();
        if (t < 64) atomicAdd(&ksum[h * M_ + c * 64 + t], msum[t]);
    }
}

// ------------------------------------------------- qp features (per-row max)
__global__ void qp_kernel(const float* __restrict__ q,
                          const float* __restrict__ proj,
                          bf16* __restrict__ qp) {
    __shared__ float rows[32][64];
    __shared__ float pr[64][65];
    __shared__ float dash[32][256];
    __shared__ float diag[32];
    __shared__ float rmax[32];
    __shared__ float red2[32][8];
    const int h = blockIdx.y;
    const int r0 = blockIdx.x * 32;
    const int t = threadIdx.x;

    for (int i = 0; i < 8; ++i) {
        int idx = t + i * 256; int rr = idx >> 6, d = idx & 63;
        rows[rr][d] = DN_ * q[((size_t)h * N_ + r0 + rr) * D_ + d];
    }
    __syncthreads();
    if (t < 32) {
        float s = 0.0f;
#pragma unroll
        for (int d = 0; d < 64; ++d) s += rows[t][d] * rows[t][d];
        diag[t] = 0.5f * s;
    }
    const int ml = t & 63, rg = t >> 6;
    for (int c = 0; c < 4; ++c) {
        __syncthreads();
        for (int i = 0; i < 16; ++i) {
            int idx = t + i * 256; int mm = idx >> 6, d = idx & 63;
            pr[mm][d] = proj[(size_t)(c * 64 + mm) * D_ + d];
        }
        __syncthreads();
        for (int rr = rg * 8; rr < rg * 8 + 8; ++rr) {
            float s = 0.0f;
#pragma unroll
            for (int d = 0; d < 64; ++d) s += rows[rr][d] * pr[ml][d];
            dash[rr][c * 64 + ml] = s;
        }
    }
    __syncthreads();
    {   // per-row max over M=256
        int row = t >> 3, seg = t & 7;
        float mx = -__builtin_inff();
        for (int j = 0; j < 32; ++j) mx = fmaxf(mx, dash[row][seg * 32 + j]);
        red2[row][seg] = mx;
    }
    __syncthreads();
    if (t < 32) {
        float mx = red2[t][0];
        for (int j = 1; j < 8; ++j) mx = fmaxf(mx, red2[t][j]);
        rmax[t] = mx;
    }
    __syncthreads();
    for (int i = 0; i < 32; ++i) {
        int idx = t + i * 256; int row = idx >> 8, m = idx & 255;
        float val = RATIO_ * (__expf(dash[row][m] - diag[row] - rmax[row]) + EPS_);
        qp[((size_t)h * N_ + r0 + row) * M_ + m] = (bf16)val;
    }
}

// ------------------------------------------------- d_inv = 1 / (qp . ksum)
__global__ void dinv_kernel(const bf16* __restrict__ qp,
                            const float* __restrict__ ksum,
                            float* __restrict__ dinv) {
    __shared__ float ks[M_];
    const int h = blockIdx.y;
    const int t = threadIdx.x;
    ks[t] = ksum[h * M_ + t];
    __syncthreads();
    const int n = blockIdx.x * 256 + t;
    const bf16* row = qp + ((size_t)h * N_ + n) * M_;
    float s = 0.0f;
#pragma unroll 8
    for (int m = 0; m < M_; ++m) s += (float)row[m] * ks[m];
    dinv[(size_t)h * N_ + n] = 1.0f / s;
}

// ------------------------------------------------- weights = qp . kp^T  (big GEMM)
// block = 256 threads (8 waves as 2x4); block tile 128x128; wave tile 64x32.
__global__ void weights_kernel(const bf16* __restrict__ qp,
                               const bf16* __restrict__ kp,
                               float* __restrict__ wout) {
    const int h = blockIdx.z;
    const int lane = threadIdx.x & 31;
    const int wave = threadIdx.x >> 5;
    const int wm = wave >> 2, wn = wave & 3;
    const int bm = blockIdx.x * 128 + wm * 64;
    const int bn = blockIdx.y * 128 + wn * 32;
    const int mrow = lane & 15;

    const bf16* qbase = qp + (size_t)h * N_ * M_;
    const bf16* kbase = kp + (size_t)h * N_ * M_;

    v8f acc[4][2] = {};
    for (int kc = 0; kc < M_; kc += 32) {
        v16bf a[4], b[2];
#pragma unroll
        for (int i = 0; i < 4; ++i)
            a[i] = load_row_frag(qbase + (size_t)(bm + i * 16 + mrow) * M_ + kc, lane);
#pragma unroll
        for (int j = 0; j < 2; ++j)
            b[j] = load_row_frag(kbase + (size_t)(bn + j * 16 + mrow) * M_ + kc, lane);
#pragma unroll
        for (int i = 0; i < 4; ++i)
#pragma unroll
            for (int j = 0; j < 2; ++j)
                acc[i][j] = __builtin_amdgcn_wmma_f32_16x16x32_bf16(
                    false, a[i], false, b[j], (short)0, acc[i][j], false, false);
    }
    const int rbase = (lane & 16) ? 8 : 0;
#pragma unroll
    for (int i = 0; i < 4; ++i)
#pragma unroll
        for (int j = 0; j < 2; ++j)
#pragma unroll
            for (int r = 0; r < 8; ++r) {
                size_t row = (size_t)bm + i * 16 + rbase + r;
                size_t col = (size_t)bn + j * 16 + (lane & 15);
                __builtin_nontemporal_store(acc[i][j][r],
                    &wout[((size_t)h * N_ + row) * N_ + col]);
            }
}

// ------------------------------------------------- context = kp^T . v  (K split)
// grid (M/16, H, 16 K-splits), block = 1 wave of 32.
__global__ void context_kernel(const bf16* __restrict__ kp,
                               const float* __restrict__ v,
                               float* __restrict__ ctx) {
    const int mt = blockIdx.x, h = blockIdx.y, ks = blockIdx.z;
    const int lane = threadIdx.x;
    const int mrow = mt * 16 + (lane & 15);
    const int koff = (lane & 16) ? 8 : 0;
    v8f acc[4] = {};
    for (int kc = ks * 256; kc < ks * 256 + 256; kc += 32) {
        v16bf a;
#pragma unroll
        for (int e = 0; e < 8; ++e)
            a[e] = kp[((size_t)h * N_ + kc + koff + e) * M_ + mrow];
#pragma unroll
        for (int e = 0; e < 8; ++e)
            a[8 + e] = kp[((size_t)h * N_ + kc + 16 + koff + e) * M_ + mrow];
#pragma unroll
        for (int j = 0; j < 4; ++j) {
            v16bf b;
            int col = j * 16 + (lane & 15);
#pragma unroll
            for (int e = 0; e < 8; ++e)
                b[e] = (bf16)v[((size_t)h * N_ + kc + koff + e) * D_ + col];
#pragma unroll
            for (int e = 0; e < 8; ++e)
                b[8 + e] = (bf16)v[((size_t)h * N_ + kc + 16 + koff + e) * D_ + col];
            acc[j] = __builtin_amdgcn_wmma_f32_16x16x32_bf16(
                false, a, false, b, (short)0, acc[j], false, false);
        }
    }
    const int rbase = (lane & 16) ? 8 : 0;
#pragma unroll
    for (int j = 0; j < 4; ++j)
#pragma unroll
        for (int r = 0; r < 8; ++r)
            atomicAdd(&ctx[((size_t)h * M_ + mt * 16 + rbase + r) * D_ + j * 16 + (lane & 15)],
                      acc[j][r]);
}

// ------------------------------------------------- out = (qp . context) * d_inv
// grid (N/16, H), block = 1 wave of 32.
__global__ void out_kernel(const bf16* __restrict__ qp,
                           const float* __restrict__ ctx,
                           const float* __restrict__ dinv,
                           float* __restrict__ outp) {
    const int nt = blockIdx.x, h = blockIdx.y;
    const int lane = threadIdx.x;
    const int koff = (lane & 16) ? 8 : 0;
    const bf16* qb = qp + ((size_t)h * N_ + nt * 16 + (lane & 15)) * M_;
    v8f acc[4] = {};
    for (int kc = 0; kc < M_; kc += 32) {
        v16bf a = load_row_frag(qb + kc, lane);
#pragma unroll
        for (int j = 0; j < 4; ++j) {
            v16bf b;
            int col = j * 16 + (lane & 15);
#pragma unroll
            for (int e = 0; e < 8; ++e)
                b[e] = (bf16)ctx[((size_t)h * M_ + kc + koff + e) * D_ + col];
#pragma unroll
            for (int e = 0; e < 8; ++e)
                b[8 + e] = (bf16)ctx[((size_t)h * M_ + kc + 16 + koff + e) * D_ + col];
            acc[j] = __builtin_amdgcn_wmma_f32_16x16x32_bf16(
                false, a, false, b, (short)0, acc[j], false, false);
        }
    }
    const int rbase = (lane & 16) ? 8 : 0;
#pragma unroll
    for (int j = 0; j < 4; ++j)
#pragma unroll
        for (int r = 0; r < 8; ++r) {
            size_t rr = (size_t)nt * 16 + rbase + r;
            float sc = dinv[(size_t)h * N_ + rr];
            outp[((size_t)h * N_ + rr) * D_ + j * 16 + (lane & 15)] = acc[j][r] * sc;
        }
}

// ---------------------------------------------------------------- launcher
extern "C" void kernel_launch(void* const* d_in, const int* in_sizes, int n_in,
                              void* d_out, int out_size, void* d_ws, size_t ws_size,
                              hipStream_t stream) {
    const float* q    = (const float*)d_in[0];
    const float* k    = (const float*)d_in[1];
    const float* v    = (const float*)d_in[2];
    const float* proj = (const float*)d_in[3];

    // Workspace layout (bytes)
    char* ws = (char*)d_ws;
    constexpr size_t QP_OFF   = 0;                                    // 32 MB bf16
    constexpr size_t KP_OFF   = QP_OFF + (size_t)H_ * N_ * M_ * 2;    // 32 MB bf16
    constexpr size_t CTX_OFF  = KP_OFF + (size_t)H_ * N_ * M_ * 2;    // 1 MB f32
    constexpr size_t KSUM_OFF = CTX_OFF + (size_t)H_ * M_ * D_ * 4;   // 16 KB f32
    constexpr size_t DINV_OFF = KSUM_OFF + (size_t)H_ * M_ * 4;       // 256 KB f32
    constexpr size_t GMAX_OFF = DINV_OFF + (size_t)H_ * N_ * 4;       // 4 B

    bf16*  qp   = (bf16*)(ws + QP_OFF);
    bf16*  kp   = (bf16*)(ws + KP_OFF);
    float* ctx  = (float*)(ws + CTX_OFF);
    float* ksum = (float*)(ws + KSUM_OFF);
    float* dinv = (float*)(ws + DINV_OFF);
    float* gmax = (float*)(ws + GMAX_OFF);

    float* out_align   = (float*)d_out;                       // [H,N,D]
    float* out_weights = (float*)d_out + (size_t)H_ * N_ * D_; // [H,N,N]

    init_kernel<<<dim3((H_ * M_ * D_ + 255) / 256), 256, 0, stream>>>(ctx, ksum, gmax);
    kdash_max_kernel<<<dim3(N_ / 64, H_), 256, 0, stream>>>(k, proj, gmax);
    kp_kernel<<<dim3(N_ / 64, H_), 256, 0, stream>>>(k, proj, gmax, kp, ksum);
    qp_kernel<<<dim3(N_ / 32, H_), 256, 0, stream>>>(q, proj, qp);
    dinv_kernel<<<dim3(N_ / 256, H_), 256, 0, stream>>>(qp, ksum, dinv);
    weights_kernel<<<dim3(N_ / 128, N_ / 128, H_), 256, 0, stream>>>(qp, kp, out_weights);
    context_kernel<<<dim3(M_ / 16, H_, 16), 32, 0, stream>>>(kp, v, ctx);
    out_kernel<<<dim3(N_ / 16, H_), 32, 0, stream>>>(qp, ctx, dinv, out_align);
}
